// ResidualLSTMKinematic_13718125543845
// MI455X (gfx1250) — matrix operand order repaired and checked by reading, not compiled, not run on hardware
//
#include <hip/hip_runtime.h>
#include <hip/hip_bf16.h>

#define B_  4096
#define T_  50
#define I_  17
#define H_  256
#define P_  20
#define G4  1024   // 4*H

typedef __attribute__((ext_vector_type(16))) __bf16 v16bf;
typedef __attribute__((ext_vector_type(8)))  __bf16 v8bf;
typedef __attribute__((ext_vector_type(8)))  float  v8f;

__device__ __forceinline__ float sigf(float x) { return 1.0f / (1.0f + __expf(-x)); }

// One 16x16x32 K-block, accumulating all four gate tiles (i,f,g,o) that share
// the same A (activation) tile.  Layouts follow CDNA5 ISA 7.12.2:
//   A 16x32 bf16 : lane m=lane&15; VGPR0-3 hold K = (hi?8:0)+0..7,
//                  VGPR4-7 hold K = 16+(hi?8:0)+0..7  -> two contiguous 8-half chunks
//   B 32x16 bf16 : lane n=lane&15; same K order as A: (hi?8:0)+0..7, then 16+(hi?8:0)+0..7
__device__ __forceinline__ void kblock4(
    const __bf16* __restrict__ Abase, int aStride,
    const __bf16* __restrict__ Wbase, int wStride,
    int k0, int bt0, int j0, int n, bool hi, v8f acc[4])
{
    const __bf16* arow = Abase + (size_t)(bt0 + n) * aStride;
    const int ka0 = k0 + (hi ? 8 : 0);
    v8bf a0 = *(const v8bf*)(arow + ka0);
    v8bf a1 = *(const v8bf*)(arow + ka0 + 16);
    v16bf a;
    #pragma unroll
    for (int e = 0; e < 8; ++e) { a[e] = a0[e]; a[8 + e] = a1[e]; }

    const int kb0 = k0 + (hi ? 8 : 0);
    #pragma unroll
    for (int g = 0; g < 4; ++g) {
        const __bf16* wrow = Wbase + (size_t)(g * H_ + j0 + n) * wStride + kb0;
        v8bf b0 = *(const v8bf*)(wrow);
        v8bf b1 = *(const v8bf*)(wrow + 16);
        v16bf b;
        #pragma unroll
        for (int e = 0; e < 8; ++e) { b[e] = b0[e]; b[8 + e] = b1[e]; }
        acc[g] = __builtin_amdgcn_wmma_f32_16x16x32_bf16(
            false, a, false, b, (short)0, acc[g], false, false);
    }
}

// One LSTM timestep for one layer, fully fused:
//   gates = hprev @ whh^T  +  ain @ win^T  + (b_ih + b_hh)
//   c = sig(f)*c + sig(i)*tanh(g) ;  h = sig(o)*tanh(c)
// Grid: (B/16, 4) x 128 threads (4 waves).  Wave w owns batch tile blockIdx.x,
// gate columns j0 = blockIdx.y*64 + w*16 (per-gate offset g*256 added inside).
__global__ __launch_bounds__(128) void lstm_step_kernel(
    const __bf16* __restrict__ hprev,                       // (B, 256) bf16
    const __bf16* __restrict__ whh,                         // (1024, 256) bf16
    const __bf16* __restrict__ ain, int ainStride, int ainKIters,
    const __bf16* __restrict__ win, int winStride,          // (1024, winStride) bf16
    const float*  __restrict__ bsum,                        // (1024) f32
    float*        __restrict__ cst,                         // (B, 256) f32, in/out
    __bf16*       __restrict__ hout)                        // (B, 256) bf16, out
{
    const int lane = threadIdx.x & 31;
    const int wave = threadIdx.x >> 5;
    const int bt0  = blockIdx.x * 16;
    const int j0   = blockIdx.y * 64 + wave * 16;
    const int n    = lane & 15;
    const bool hi  = lane >= 16;

    v8f acc[4];
    #pragma unroll
    for (int g = 0; g < 4; ++g) acc[g] = 0.0f;

    // recurrent part: h_prev (K = 256)
    #pragma unroll
    for (int kk = 0; kk < 8; ++kk)
        kblock4(hprev, H_, whh, H_, kk * 32, bt0, j0, n, hi, acc);

    // input-projection part: layer0 -> padded x (1 block), layer1 -> h0 (8 blocks)
    for (int kk = 0; kk < ainKIters; ++kk)
        kblock4(ain, ainStride, win, winStride, kk * 32, bt0, j0, n, hi, acc);

    float bj[4];
    #pragma unroll
    for (int g = 0; g < 4; ++g) bj[g] = bsum[g * H_ + j0 + n];

    // C/D layout: VGPR r, lanes<16 -> row r, lanes>=16 -> row r+8; col = lane&15
    #pragma unroll
    for (int r = 0; r < 8; ++r) {
        const int m   = r + (hi ? 8 : 0);
        const int idx = (bt0 + m) * H_ + (j0 + n);
        const float iv = sigf(acc[0][r] + bj[0]);
        const float fv = sigf(acc[1][r] + bj[1]);
        const float gv = tanhf(acc[2][r] + bj[2]);
        const float ov = sigf(acc[3][r] + bj[3]);
        const float cn = fv * cst[idx] + iv * gv;
        cst[idx]  = cn;
        hout[idx] = (__bf16)(ov * tanhf(cn));
    }
}

// Weight/bias conversion (runs every call; graph-replay safe).
__global__ void prep_weights(
    const float* __restrict__ wih0, const float* __restrict__ whh0,
    const float* __restrict__ wih1, const float* __restrict__ whh1,
    const float* __restrict__ bih0, const float* __restrict__ bhh0,
    const float* __restrict__ bih1, const float* __restrict__ bhh1,
    __bf16* __restrict__ wih0p, __bf16* __restrict__ whh0b,
    __bf16* __restrict__ wih1b, __bf16* __restrict__ whh1b,
    float* __restrict__ bsum0, float* __restrict__ bsum1)
{
    const int idx = blockIdx.x * blockDim.x + threadIdx.x;
    if (idx < G4 * H_) {
        whh0b[idx] = (__bf16)whh0[idx];
        wih1b[idx] = (__bf16)wih1[idx];
        whh1b[idx] = (__bf16)whh1[idx];
    }
    if (idx < G4 * 32) {
        const int r = idx >> 5, k = idx & 31;
        wih0p[idx] = (__bf16)(k < I_ ? wih0[r * I_ + k] : 0.0f);
    }
    if (idx < G4) {
        bsum0[idx] = bih0[idx] + bhh0[idx];
        bsum1[idx] = bih1[idx] + bhh1[idx];
    }
}

// x (B,T,17) f32 -> (B,T,32) bf16 zero-padded along K.
__global__ void convert_x(const float* __restrict__ x, __bf16* __restrict__ xb)
{
    const int idx = blockIdx.x * blockDim.x + threadIdx.x;
    if (idx >= B_ * T_ * 32) return;
    const int k  = idx & 31;
    const int bt = idx >> 5;                 // = b*T + t
    xb[idx] = (__bf16)(k < I_ ? x[bt * I_ + k] : 0.0f);
}

__global__ void zero_state(float* __restrict__ c0, float* __restrict__ c1,
                           __bf16* __restrict__ h0p, __bf16* __restrict__ h1p)
{
    const int idx = blockIdx.x * blockDim.x + threadIdx.x;
    if (idx < B_ * H_) {
        c0[idx] = 0.0f; c1[idx] = 0.0f;
        h0p[idx] = (__bf16)0.0f; h1p[idx] = (__bf16)0.0f;
    }
}

// FC head + softplus + kinematic rollout.  One thread per batch element.
__global__ __launch_bounds__(64) void fc_kin_kernel(
    const __bf16* __restrict__ hlast,
    const float* __restrict__ fcw, const float* __restrict__ fcb,
    const float* __restrict__ x, float* __restrict__ out)
{
    const int b = blockIdx.x * blockDim.x + threadIdx.x;
    if (b >= B_) return;
    const __bf16* h = hlast + (size_t)b * H_;
    float raw[2 * P_];
    for (int rr = 0; rr < 2 * P_; ++rr) {
        float s = fcb[rr];
        const float* w = fcw + (size_t)rr * H_;
        for (int k = 0; k < H_; ++k) s += (float)h[k] * w[k];
        raw[rr] = s;
    }
    const float DTc  = 0.1f;
    const float IMGW = 1920.0f, IMGH = 1080.0f;
    const float DIAG = 2202.907170f;
    const size_t xoff = (size_t)b * T_ * I_ + (size_t)(T_ - 1) * I_;
    const float x0 = x[xoff + 0], y0 = x[xoff + 1];
    float psi = atan2f(x[xoff + 7], x[xoff + 8]);
    float cx = x0 * IMGW, cy = y0 * IMGH;
    float decay = 1.0f;
    float* ob = out + (size_t)b * P_ * 2;
    for (int p = 0; p < P_; ++p) {
        const float rs    = raw[2 * p];
        const float speed = (rs > 20.0f) ? rs : log1pf(__expf(rs));
        const float w     = raw[2 * p + 1] * decay;
        const float v     = speed * DIAG;
        const float psiA  = psi + w * DTc;
        const bool  straight = fabsf(w) < 0.01f;
        const float wsafe = w + (straight ? 1e-4f : 0.0f);
        const float radius = v / wsafe;
        const float dx = straight ? v * __cosf(psi) * DTc
                                  : radius * (__sinf(psiA) - __sinf(psi));
        const float dy = straight ? v * __sinf(psi) * DTc
                                  : -radius * (__cosf(psiA) - __cosf(psi));
        cx += dx; cy += dy;
        ob[2 * p]     = cx / IMGW;
        ob[2 * p + 1] = cy / IMGH;
        psi = psiA;
        decay *= 0.97f;
    }
}

extern "C" void kernel_launch(void* const* d_in, const int* in_sizes, int n_in,
                              void* d_out, int out_size, void* d_ws, size_t ws_size,
                              hipStream_t stream)
{
    (void)in_sizes; (void)n_in; (void)out_size; (void)ws_size;
    const float* x    = (const float*)d_in[0];
    const float* wih0 = (const float*)d_in[1];
    const float* whh0 = (const float*)d_in[2];
    const float* bih0 = (const float*)d_in[3];
    const float* bhh0 = (const float*)d_in[4];
    const float* wih1 = (const float*)d_in[5];
    const float* whh1 = (const float*)d_in[6];
    const float* bih1 = (const float*)d_in[7];
    const float* bhh1 = (const float*)d_in[8];
    const float* fcw  = (const float*)d_in[9];
    const float* fcb  = (const float*)d_in[10];
    float* out = (float*)d_out;

    char* ws = (char*)d_ws;
    size_t off = 0;
    auto wsAlloc = [&](size_t bytes) {
        char* p = ws + off;
        off += (bytes + 255) & ~(size_t)255;
        return p;
    };
    __bf16* xb    = (__bf16*)wsAlloc((size_t)B_ * T_ * 32 * 2);   // padded x, bf16
    __bf16* wih0p = (__bf16*)wsAlloc((size_t)G4 * 32 * 2);        // w_ih0 K-padded
    __bf16* whh0b = (__bf16*)wsAlloc((size_t)G4 * H_ * 2);
    __bf16* wih1b = (__bf16*)wsAlloc((size_t)G4 * H_ * 2);
    __bf16* whh1b = (__bf16*)wsAlloc((size_t)G4 * H_ * 2);
    float*  bsum0 = (float*)wsAlloc((size_t)G4 * 4);
    float*  bsum1 = (float*)wsAlloc((size_t)G4 * 4);
    float*  c0    = (float*)wsAlloc((size_t)B_ * H_ * 4);
    float*  c1    = (float*)wsAlloc((size_t)B_ * H_ * 4);
    __bf16* h0a   = (__bf16*)wsAlloc((size_t)B_ * H_ * 2);
    __bf16* h0b   = (__bf16*)wsAlloc((size_t)B_ * H_ * 2);
    __bf16* h1a   = (__bf16*)wsAlloc((size_t)B_ * H_ * 2);
    __bf16* h1b   = (__bf16*)wsAlloc((size_t)B_ * H_ * 2);

    prep_weights<<<(G4 * H_ + 255) / 256, 256, 0, stream>>>(
        wih0, whh0, wih1, whh1, bih0, bhh0, bih1, bhh1,
        wih0p, whh0b, wih1b, whh1b, bsum0, bsum1);
    convert_x<<<(B_ * T_ * 32 + 255) / 256, 256, 0, stream>>>(x, xb);
    zero_state<<<(B_ * H_ + 255) / 256, 256, 0, stream>>>(c0, c1, h0b, h1b);

    dim3 grid(B_ / 16, 4), blk(128);
    for (int t = 0; t < T_; ++t) {
        __bf16*       h0cur  = (t & 1) ? h0b : h0a;
        const __bf16* h0prev = (t & 1) ? h0a : h0b;
        lstm_step_kernel<<<grid, blk, 0, stream>>>(
            h0prev, whh0b, xb + (size_t)t * 32, T_ * 32, 1, wih0p, 32,
            bsum0, c0, h0cur);
        __bf16*       h1cur  = (t & 1) ? h1b : h1a;
        const __bf16* h1prev = (t & 1) ? h1a : h1b;
        lstm_step_kernel<<<grid, blk, 0, stream>>>(
            h1prev, whh1b, h0cur, H_, 8, wih1b, H_,
            bsum1, c1, h1cur);
    }
    const __bf16* hlast = ((T_ - 1) & 1) ? h1b : h1a;   // = h1b for T=50
    fc_kin_kernel<<<(B_ + 63) / 64, 64, 0, stream>>>(hlast, fcw, fcb, x, out);
}
